// SharedWeightFC_38895223832848
// MI455X (gfx1250) — matrix operand order repaired and checked by reading, not compiled, not run on hardware
//
#include <hip/hip_runtime.h>

// ---------------------------------------------------------------------------
// y[b, o] = dot(x[b,:], shared_weights[o % 256, :]) + bias[o]
// Compact GEMM [8192 x 4096] x [4096 x 256]^T -> [8192 x 256], broadcast x16
// with bias in the epilogue. bf16 WMMA 16x16x32, f32 accumulate.
// Software pipeline: global->VGPR (tile k+2) || convert->LDS (tile k+1,
// ping-pong buffer) || WMMA from LDS (tile k). One barrier per K-step.
// This round: LDS-only barrier (s_wait_dscnt + s_barrier) instead of
// __syncthreads(), so the in-flight global loads for tile k+2 are NOT drained
// at the barrier -> full-iteration HBM latency hiding.
// ---------------------------------------------------------------------------

typedef __attribute__((ext_vector_type(16))) __bf16        v16bf;
typedef __attribute__((ext_vector_type(8)))  float         v8f;
typedef __attribute__((ext_vector_type(4)))  float         v4f;
typedef __attribute__((ext_vector_type(4)))  unsigned int  v4u;
typedef __attribute__((ext_vector_type(2)))  unsigned int  v2u;

#define KDIM     4096
#define NOUT     4096
#define NSHARE   256
#define BATCH    8192
#define BM       64          // rows per workgroup
#define BN       128         // compact cols per workgroup (2 N-splits of 256)
#define BK       32          // K per WMMA step (bf16 16x16x32)
#define PITCH    (BK + 8)    // LDS row pitch in bf16 elems (80B: staggered banks)
#define NTHREADS 256         // 8 waves (wave32)

union ABfrag { v4u u[2]; v16bf v; };

// f32 pair -> packed bf16. Single HW instruction on gfx1250 (src0->lo, src1->hi).
static __device__ __forceinline__ unsigned int pk2bf(float a, float b) {
  unsigned int u;
  asm("v_cvt_pk_bf16_f32 %0, %1, %2" : "=v"(u) : "v"(a), "v"(b));
  return u;
}

// Workgroup barrier that only drains LDS (DScnt), NOT vector-memory loadcnt.
// Cross-wave visibility through the barrier only needs the ds_stores complete;
// global loads land in private VGPRs and are ordered by their register uses.
static __device__ __forceinline__ void wg_barrier_lds() {
  asm volatile("s_wait_dscnt 0x0" ::: "memory");
  __builtin_amdgcn_s_barrier();
  asm volatile("" ::: "memory");
}

__global__ __launch_bounds__(NTHREADS)
void shared_fc_wmma_bf16(const float* __restrict__ x,
                         const float* __restrict__ w,
                         const float* __restrict__ bias,
                         float* __restrict__ out)
{
  __shared__ unsigned short xT[2][BM * PITCH];   // 2 x 5 KB
  __shared__ unsigned short wT[2][BN * PITCH];   // 2 x 10 KB
  __shared__ float          bS[16 * BN];         // 8 KB bias slice

  const int tid   = threadIdx.x;
  const int lane  = tid & 31;
  const int wave  = tid >> 5;
  const int laneM = lane & 15;   // M (A) / N (B) index within fragment
  const int laneH = lane >> 4;   // half-wave selector
  const int wm    = wave & 3;    // 4 m-subtiles of 16 rows
  const int wn    = wave >> 2;   // 2 n-groups of 64 cols
  const int mBase = blockIdx.x * BM;
  const int nBase = blockIdx.y * BN;

  // stage the bias entries this split will emit: o = rep*256 + nBase + j
  for (int i = tid; i < 16 * BN; i += NTHREADS) {
    const int rep = i >> 7;
    const int j   = i & (BN - 1);
    bS[i] = bias[rep * NSHARE + nBase + j];
  }

  // staging geometry: thread covers column slice c4 of rows sRow + 32*i
  const int sRow = tid >> 3;            // 0..31
  const int c4   = (tid & 7) << 2;      // 0,4,..,28

  v4f gx[2];   // x rows sRow, sRow+32
  v4f gw[4];   // w rows sRow, +32, +64, +96

  auto issue_loads = [&](int k0) {
#pragma unroll
    for (int i = 0; i < 2; ++i)
      gx[i] = *(const v4f*)(x + (size_t)(mBase + sRow + 32 * i) * KDIM + k0 + c4);
#pragma unroll
    for (int i = 0; i < 4; ++i)
      gw[i] = *(const v4f*)(w + (size_t)(nBase + sRow + 32 * i) * KDIM + k0 + c4);
  };

  auto convert_store = [&](int buf) {
#pragma unroll
    for (int i = 0; i < 2; ++i) {
      v2u p; p.x = pk2bf(gx[i].x, gx[i].y); p.y = pk2bf(gx[i].z, gx[i].w);
      *(v2u*)&xT[buf][(sRow + 32 * i) * PITCH + c4] = p;
    }
#pragma unroll
    for (int i = 0; i < 4; ++i) {
      v2u p; p.x = pk2bf(gw[i].x, gw[i].y); p.y = pk2bf(gw[i].z, gw[i].w);
      *(v2u*)&wT[buf][(sRow + 32 * i) * PITCH + c4] = p;
    }
  };

  v8f acc[4] = {};

  // ---- pipeline prologue ------------------------------------------------
  issue_loads(0);
  convert_store(0);
  issue_loads(BK);          // tile 1 in flight during first compute
  wg_barrier_lds();

  // ---- main loop: 1 barrier per K-step, unroll 2 for static ping-pong --
#pragma unroll 2
  for (int k0 = 0; k0 < KDIM; k0 += BK) {
    const int cur = (k0 >> 5) & 1;

    // A fragment (16-bit A 16x32 layout):
    // lanes 0-15: K 0-7 (v0-3), K 16-23 (v4-7); lanes 16-31: +8
    ABfrag a;
    {
      const unsigned short* p = &xT[cur][(wm * 16 + laneM) * PITCH + laneH * 8];
      a.u[0] = *(const v4u*)p;
      a.u[1] = *(const v4u*)(p + 16);
    }
    // B fragments (B 32x16: lanes 0-15 hold K 0-7, 16-23; lanes 16-31: +8)
    // then the WMMA burst; scheduler overlaps ds_loads with XDL execution.
    ABfrag b[4];
#pragma unroll
    for (int s = 0; s < 4; ++s) {
      const unsigned short* p =
          &wT[cur][(wn * 64 + s * 16 + laneM) * PITCH + laneH * 8];
      b[s].u[0] = *(const v4u*)p;
      b[s].u[1] = *(const v4u*)(p + 16);
    }
#pragma unroll
    for (int s = 0; s < 4; ++s) {
      acc[s] = __builtin_amdgcn_wmma_f32_16x16x32_bf16(
          false, a.v, false, b[s].v, (short)0, acc[s], false, false);
    }

    // stage tile k+1 into the other buffer; launch loads for tile k+2
    if (k0 + BK < KDIM) {
      convert_store(cur ^ 1);               // waits on loads issued last iter
      if (k0 + 2 * BK < KDIM) issue_loads(k0 + 2 * BK);
    }
    wg_barrier_lds();
  }

  // ---- epilogue: broadcast compact cols to 16 repeats + bias ------------
  // C layout: vgpr r holds M = laneH*8 + r, N = laneM
#pragma unroll
  for (int s = 0; s < 4; ++s) {
    const int colLoc = wn * 64 + s * 16 + laneM;   // 0..127 within split
    float bv[16];
#pragma unroll
    for (int rep = 0; rep < 16; ++rep) bv[rep] = bS[rep * BN + colLoc];
#pragma unroll
    for (int r = 0; r < 8; ++r) {
      const float v = acc[s][r];
      float* orow = out + (size_t)(mBase + wm * 16 + laneH * 8 + r) * NOUT
                        + nBase + colLoc;
#pragma unroll 4
      for (int rep = 0; rep < 16; ++rep) {
        __builtin_nontemporal_store(v + bv[rep], orow + rep * NSHARE);
      }
    }
  }
}

extern "C" void kernel_launch(void* const* d_in, const int* in_sizes, int n_in,
                              void* d_out, int out_size, void* d_ws, size_t ws_size,
                              hipStream_t stream) {
  const float* x  = (const float*)d_in[0];   // [8192, 4096]
  const float* w  = (const float*)d_in[1];   // [256, 4096]
  const float* b  = (const float*)d_in[2];   // [4096]
  float* out      = (float*)d_out;           // [8192, 4096]

  dim3 grid(BATCH / BM, NSHARE / BN);        // (128, 2)
  shared_fc_wmma_bf16<<<grid, dim3(NTHREADS), 0, stream>>>(x, w, b, out);
}
